// TransformerEncoderLayer_86706799772197
// MI455X (gfx1250) — matrix operand order repaired and checked by reading, not compiled, not run on hardware
//
#include <hip/hip_runtime.h>
#include <hip/hip_bf16.h>
#include <stdint.h>

// MI455X / gfx1250, wave32. All matmuls via v_wmma_f32_16x16x32_bf16.
// Fragments are built with aligned 16B vector loads; attention uses online
// (flash) softmax so nothing spills.

typedef __attribute__((ext_vector_type(16))) __bf16 v16bf;
typedef __attribute__((ext_vector_type(8)))  float  v8f;

union F16x16 {
  unsigned short u[16];
  uint4 q4[2];
  v16bf v;
};
union U4x8 {
  uint4 q;
  unsigned short u[8];
};

__device__ __forceinline__ unsigned short f2bf(float f) {
  unsigned int u = __float_as_uint(f);
  u += 0x7FFFu + ((u >> 16) & 1u);            // round-to-nearest-even
  return (unsigned short)(u >> 16);
}
__device__ __forceinline__ float qgelu(float x) {
  return x / (1.f + __expf(-1.702f * x));
}
__device__ __forceinline__ void wait_ds0() {
#if __has_builtin(__builtin_amdgcn_s_wait_dscnt)
  __builtin_amdgcn_s_wait_dscnt(0);
#else
  asm volatile("s_wait_dscnt 0" ::: "memory");
#endif
}

// ---------------------------------------------------------------------------
// Generic WMMA GEMM: out = act(A*B + bias) [+ resid]
//   A: MxK row-major (f32 or bf16), B: KxN row-major f32 (converted to bf16
//   while staging). K % 32 == 0, N % 64 == 0.
// Block 256 threads (8 waves, 4x2); BM=128, BN=64, BK=32; wave tile 32x32.
// B tile staged TRANSPOSED (Bst[n][k]) so B-fragments are contiguous b128s.
// OUT_MODE: 0 = f32 row-major, 1 = bf16 row-major, 2 = bf16 QKV layout
//   ((s*12 + h)*Lp + t)*64 + c  with row = s*197 + t, col = h*64 + c.
// ---------------------------------------------------------------------------
template<bool A_BF16, int ACT, int OUT_MODE, bool RESID>
__global__ __launch_bounds__(256) void gemm_k(
    const void* __restrict__ Ap, const float* __restrict__ B,
    const float* __restrict__ bias, const float* __restrict__ resid,
    void* __restrict__ outp, int M, int N, int K, int Lp)
{
  __shared__ __align__(16) unsigned short As[128][40];   // row-major, +8 pad
  __shared__ __align__(16) unsigned short Bst[64][40];   // transposed, +8 pad

  const int tid  = threadIdx.x;
  const int bm   = blockIdx.y * 128;
  const int bn   = blockIdx.x * 64;
  const int w    = tid >> 5;
  const int lane = tid & 31;
  const int lo   = lane & 15, hi = lane >> 4;
  const int wm   = (w >> 1) * 32;   // 0,32,64,96
  const int wn   = (w & 1) * 32;    // 0,32

  const unsigned short* A16 = (const unsigned short*)Ap;
  const float*          A32 = (const float*)Ap;

  const int ar = tid >> 1, ac = (tid & 1) * 16;   // A stage: 16 elems/thread
  const int br = tid >> 2, bc = (tid & 3) * 16;   // B stage: 16 elems/thread
  const int agrow = bm + ar;

  v8f acc[2][2] = {};

  for (int k0 = 0; k0 < K; k0 += 32) {
    // ---- stage A tile 128x32 (two aligned 16B LDS writes per thread)
    {
      F16x16 tmp;
      if (agrow < M) {
        if (A_BF16) {
          const uint4* src = (const uint4*)(A16 + (size_t)agrow * K + k0 + ac);
          tmp.q4[0] = src[0];
          tmp.q4[1] = src[1];
        } else {
          const float4* src = (const float4*)(A32 + (size_t)agrow * K + k0 + ac);
          #pragma unroll
          for (int j = 0; j < 4; ++j) {
            float4 fv = src[j];
            tmp.u[4 * j + 0] = f2bf(fv.x);
            tmp.u[4 * j + 1] = f2bf(fv.y);
            tmp.u[4 * j + 2] = f2bf(fv.z);
            tmp.u[4 * j + 3] = f2bf(fv.w);
          }
        }
      } else {
        tmp.q4[0] = make_uint4(0, 0, 0, 0);
        tmp.q4[1] = make_uint4(0, 0, 0, 0);
      }
      *(uint4*)&As[ar][ac]     = tmp.q4[0];
      *(uint4*)&As[ar][ac + 8] = tmp.q4[1];
      // prefetch next A K-tile
      if (k0 + 32 < K && agrow < M) {
        const char* nxt = A_BF16
            ? (const char*)(A16 + (size_t)agrow * K + k0 + 32 + ac)
            : (const char*)(A32 + (size_t)agrow * K + k0 + 32 + ac);
        __builtin_prefetch(nxt, 0, 3);
      }
    }
    // ---- stage B tile 32x64 transposed (first 128 threads)
    if (tid < 128) {
      const float4* src = (const float4*)(B + (size_t)(k0 + br) * N + bn + bc);
      #pragma unroll
      for (int j = 0; j < 4; ++j) {
        float4 fv = src[j];
        Bst[bc + 4 * j + 0][br] = f2bf(fv.x);
        Bst[bc + 4 * j + 1][br] = f2bf(fv.y);
        Bst[bc + 4 * j + 2][br] = f2bf(fv.z);
        Bst[bc + 4 * j + 3][br] = f2bf(fv.w);
      }
      if (k0 + 32 < K)
        __builtin_prefetch((const char*)(B + (size_t)(k0 + 32 + br) * N + bn + bc), 0, 3);
    }
    __syncthreads();

    // ---- fragments: aligned b128 LDS reads
    F16x16 af[2], bf2[2];
    #pragma unroll
    for (int mt = 0; mt < 2; ++mt) {
      int r = wm + mt * 16 + lo;
      af[mt].q4[0] = *(const uint4*)&As[r][16 * hi];       // K = 16*hi + 0..7
      af[mt].q4[1] = *(const uint4*)&As[r][16 * hi + 8];   // K = 16*hi + 8..15
    }
    #pragma unroll
    for (int nt = 0; nt < 2; ++nt) {
      int c = wn + nt * 16 + lo;
      bf2[nt].q4[0] = *(const uint4*)&Bst[c][16 * hi];     // K = 16*hi + 0..7
      bf2[nt].q4[1] = *(const uint4*)&Bst[c][16 * hi + 8]; // K = 16*hi + 8..15
    }
    #pragma unroll
    for (int mt = 0; mt < 2; ++mt)
      #pragma unroll
      for (int nt = 0; nt < 2; ++nt)
        acc[mt][nt] = __builtin_amdgcn_wmma_f32_16x16x32_bf16(
            false, af[mt].v, false, bf2[nt].v, (short)0, acc[mt][nt],
            false, false);
    __syncthreads();
  }

  // ---- epilogue (D layout: VGPR r, lane half -> m = r + 8*hi, n = lo)
  float*          of = (float*)outp;
  unsigned short* oh = (unsigned short*)outp;
  #pragma unroll
  for (int mt = 0; mt < 2; ++mt) {
    #pragma unroll
    for (int nt = 0; nt < 2; ++nt) {
      int col = bn + wn + nt * 16 + lo;
      #pragma unroll
      for (int r = 0; r < 8; ++r) {
        int row = bm + wm + mt * 16 + r + 8 * hi;
        if (row < M) {
          float val = acc[mt][nt][r] + bias[col];
          if (ACT == 1) val = qgelu(val);
          if (RESID)    val += resid[(size_t)row * N + col];
          if (OUT_MODE == 0) {
            of[(size_t)row * N + col] = val;
          } else if (OUT_MODE == 1) {
            oh[(size_t)row * N + col] = f2bf(val);
          } else { // QKV scatter
            int s = row / 197, t = row - s * 197;
            int hh = col >> 6, cc = col & 63;
            oh[(((size_t)s * 12 + hh) * Lp + t) * 64 + cc] = f2bf(val);
          }
        }
      }
    }
  }
}

// ---------------------------------------------------------------------------
// Flash attention: one block per (sample, head); q/k/v bf16 (sh, Lp=224, 64),
// rows t in [197,224) zero-padded.  Per wave: 16 query rows; keys processed
// in pairs of 16 (K=32 per WMMA) with online softmax -> only 2 live score
// accumulators + 4 output accumulators (no spills).  V staged transposed in
// LDS once per block; P bounced via per-wave LDS (D-layout -> A-layout).
// ---------------------------------------------------------------------------
__global__ __launch_bounds__(256) void attn_k(
    const unsigned short* __restrict__ q, const unsigned short* __restrict__ k,
    const unsigned short* __restrict__ v, unsigned short* __restrict__ mix)
{
  const int sh = blockIdx.x;                 // s*12 + h
  const int s = sh / 12, h = sh - s * 12;
  const size_t base = (size_t)sh * (224 * 64);
  const int tid = threadIdx.x;
  const int w = tid >> 5, lane = tid & 31, lo = lane & 15, hi = lane >> 4;

  __shared__ __align__(16) unsigned short Vt[64][224];      // V transposed
  __shared__ __align__(16) unsigned short Pst[8][16][32];   // per-wave P pair

  // stage V transposed: Vt[c][t] = v[t][c]
  for (int idx = tid; idx < 224 * 64 / 8; idx += 256) {
    int t  = idx >> 3;
    int c8 = (idx & 7) * 8;
    U4x8 d;
    d.q = *(const uint4*)(v + base + (size_t)t * 64 + c8);
    #pragma unroll
    for (int j = 0; j < 8; ++j) Vt[c8 + j][t] = d.u[j];
  }
  __syncthreads();

  for (int qt = w; qt < 14; qt += 8) {
    // Q A-fragments (K = 64 -> 2 steps of 32), contiguous b128 global reads
    F16x16 qa[2];
    const size_t qrow = base + (size_t)(qt * 16 + lo) * 64;
    #pragma unroll
    for (int kk = 0; kk < 2; ++kk) {
      qa[kk].q4[0] = *(const uint4*)(q + qrow + kk * 32 + 16 * hi);
      qa[kk].q4[1] = *(const uint4*)(q + qrow + kk * 32 + 16 * hi + 8);
    }

    float rm[8], rl[8], alpha[8];
    #pragma unroll
    for (int r = 0; r < 8; ++r) { rm[r] = -3.0e38f; rl[r] = 0.f; }
    v8f macc[4] = {};

    for (int kp = 0; kp < 7; ++kp) {        // key pair: tokens [kp*32, kp*32+32)
      // scores for two 16-wide key tiles
      v8f s0 = {}, s1 = {};
      #pragma unroll
      for (int kk = 0; kk < 2; ++kk) {
        F16x16 kb0, kb1;
        const size_t kr0 = base + (size_t)(kp * 32 + lo) * 64 + kk * 32 + 16 * hi;
        const size_t kr1 = base + (size_t)(kp * 32 + 16 + lo) * 64 + kk * 32 + 16 * hi;
        kb0.q4[0] = *(const uint4*)(k + kr0);
        kb0.q4[1] = *(const uint4*)(k + kr0 + 8);
        kb1.q4[0] = *(const uint4*)(k + kr1);
        kb1.q4[1] = *(const uint4*)(k + kr1 + 8);
        s0 = __builtin_amdgcn_wmma_f32_16x16x32_bf16(
            false, qa[kk].v, false, kb0.v, (short)0, s0, false, false);
        s1 = __builtin_amdgcn_wmma_f32_16x16x32_bf16(
            false, qa[kk].v, false, kb1.v, (short)0, s1, false, false);
      }

      // online softmax update (rows r + 8*hi within wave's 16 q rows)
      const int n0 = kp * 32 + lo, n1 = kp * 32 + 16 + lo;
      #pragma unroll
      for (int r = 0; r < 8; ++r) {
        float v0 = (n0 < 197) ? s0[r] * 0.125f : -3.0e38f;
        float v1 = (n1 < 197) ? s1[r] * 0.125f : -3.0e38f;
        float tm = fmaxf(v0, v1);
        #pragma unroll
        for (int m = 1; m < 16; m <<= 1) tm = fmaxf(tm, __shfl_xor(tm, m, 32));
        float Mn = fmaxf(rm[r], tm);
        float a  = __expf(rm[r] - Mn);
        float p0 = __expf(v0 - Mn);
        float p1 = __expf(v1 - Mn);
        float ts = p0 + p1;
        #pragma unroll
        for (int m = 1; m < 16; m <<= 1) ts += __shfl_xor(ts, m, 32);
        rl[r] = rl[r] * a + ts;
        rm[r] = Mn;
        alpha[r] = a;
        Pst[w][r + 8 * hi][lo]      = f2bf(p0);
        Pst[w][r + 8 * hi][16 + lo] = f2bf(p1);
      }
      wait_ds0();

      // P A-fragment (16x32) from per-wave LDS, aligned b128 reads
      F16x16 pa;
      pa.q4[0] = *(const uint4*)&Pst[w][lo][16 * hi];
      pa.q4[1] = *(const uint4*)&Pst[w][lo][16 * hi + 8];

      #pragma unroll
      for (int nt = 0; nt < 4; ++nt) {
        #pragma unroll
        for (int r = 0; r < 8; ++r) macc[nt][r] *= alpha[r];   // rescale
        F16x16 vb;   // V B-frag: K = kp*32 + 16*hi + 0..15, contiguous in Vt
        vb.q4[0] = *(const uint4*)&Vt[nt * 16 + lo][kp * 32 + 16 * hi];
        vb.q4[1] = *(const uint4*)&Vt[nt * 16 + lo][kp * 32 + 16 * hi + 8];
        macc[nt] = __builtin_amdgcn_wmma_f32_16x16x32_bf16(
            false, pa.v, false, vb.v, (short)0, macc[nt], false, false);
      }
    }

    // normalize and store (token-major bf16 for the WO GEMM)
    #pragma unroll
    for (int nt = 0; nt < 4; ++nt)
      #pragma unroll
      for (int r = 0; r < 8; ++r) {
        int t = qt * 16 + r + 8 * hi;
        if (t < 197)
          mix[((size_t)(s * 197 + t)) * 768 + h * 64 + nt * 16 + lo] =
              f2bf(macc[nt][r] / rl[r]);
      }
  }
}

// ---------------------------------------------------------------------------
// Elementwise / reduction helpers
// ---------------------------------------------------------------------------
__global__ __launch_bounds__(256) void zero_k(unsigned int* p, size_t nwords) {
  size_t i = (size_t)blockIdx.x * 256 + threadIdx.x;
  if (i < nwords) p[i] = 0u;
}

// depthwise temporal conv (k=3, pad=1 over T=8) + quickGELU
__global__ __launch_bounds__(256) void convgelu_k(
    const float* __restrict__ pd, const float* __restrict__ wconv,
    const float* __restrict__ bconv, unsigned short* __restrict__ pg)
{
  size_t g = (size_t)blockIdx.x * 256 + threadIdx.x;
  if (g >= (size_t)12544 * 384) return;
  int ch = (int)(g % 384);
  size_t r = g / 384;
  int tok = (int)(r % 196);
  int s   = (int)(r / 196);
  int b = s >> 3, t = s & 7;                    // T = 8
  float acc = bconv[ch];
  #pragma unroll
  for (int kk = 0; kk < 3; ++kk) {
    int tt = t + kk - 1;
    if (tt >= 0 && tt < 8)
      acc += wconv[ch * 3 + kk] *
             pd[((size_t)(b * 8 + tt) * 197 + 1 + tok) * 384 + ch];
  }
  pg[g] = f2bf(qgelu(acc));
}

// x1 = x + concat([cls, pu])
__global__ __launch_bounds__(256) void addmsg_k(
    const float* __restrict__ x, const float* __restrict__ pu,
    float* __restrict__ x1)
{
  size_t g = (size_t)blockIdx.x * 256 + threadIdx.x;
  if (g >= (size_t)12608 * 768) return;
  int d = (int)(g % 768);
  size_t rt = g / 768;
  int s = (int)(rt / 197), t = (int)(rt % 197);
  float xv = x[g];
  float add = (t == 0) ? xv : pu[((size_t)s * 196 + (t - 1)) * 768 + d];
  x1[g] = xv + add;
}

// row LayerNorm over 768 -> bf16
__global__ __launch_bounds__(256) void ln_k(
    const float* __restrict__ x, const float* __restrict__ g,
    const float* __restrict__ b, unsigned short* __restrict__ out)
{
  int row = blockIdx.x;
  const float* xr = x + (size_t)row * 768;
  int tid = threadIdx.x;
  float vals[3], s = 0.f, s2 = 0.f;
  #pragma unroll
  for (int i = 0; i < 3; ++i) {
    float vv = xr[tid + 256 * i];
    vals[i] = vv; s += vv; s2 += vv * vv;
  }
  #pragma unroll
  for (int m = 1; m < 32; m <<= 1) {
    s  += __shfl_xor(s,  m, 32);
    s2 += __shfl_xor(s2, m, 32);
  }
  __shared__ float rs[8], rs2[8];
  int w = tid >> 5, lane = tid & 31;
  if (lane == 0) { rs[w] = s; rs2[w] = s2; }
  __syncthreads();
  float ts = 0.f, ts2 = 0.f;
  #pragma unroll
  for (int i = 0; i < 8; ++i) { ts += rs[i]; ts2 += rs2[i]; }
  float mean = ts * (1.f / 768.f);
  float var  = ts2 * (1.f / 768.f) - mean * mean;
  float rstd = rsqrtf(var + 1e-5f);
  #pragma unroll
  for (int i = 0; i < 3; ++i) {
    int c = tid + 256 * i;
    out[(size_t)row * 768 + c] = f2bf((vals[i] - mean) * rstd * g[c] + b[c]);
  }
}

// ---------------------------------------------------------------------------
extern "C" void kernel_launch(void* const* d_in, const int* in_sizes, int n_in,
                              void* d_out, int out_size, void* d_ws,
                              size_t ws_size, hipStream_t stream)
{
  (void)in_sizes; (void)n_in; (void)out_size; (void)ws_size;
  const float* x      = (const float*)d_in[0];
  const float* w_down = (const float*)d_in[1];
  const float* b_down = (const float*)d_in[2];
  const float* w_conv = (const float*)d_in[3];
  const float* b_conv = (const float*)d_in[4];
  const float* w_up   = (const float*)d_in[5];
  const float* b_up   = (const float*)d_in[6];
  const float* g1     = (const float*)d_in[7];
  const float* be1    = (const float*)d_in[8];
  const float* wq     = (const float*)d_in[9];
  const float* bq     = (const float*)d_in[10];
  const float* wkp    = (const float*)d_in[11];
  const float* bk     = (const float*)d_in[12];
  const float* wvp    = (const float*)d_in[13];
  const float* bv     = (const float*)d_in[14];
  const float* wo     = (const float*)d_in[15];
  const float* bo     = (const float*)d_in[16];
  const float* g2     = (const float*)d_in[17];
  const float* be2    = (const float*)d_in[18];
  const float* w1     = (const float*)d_in[19];
  const float* b1     = (const float*)d_in[20];
  const float* w2     = (const float*)d_in[21];
  const float* b2     = (const float*)d_in[22];
  float* out = (float*)d_out;

  constexpr int Dm = 768, DHm = 384, MLPD = 3072, Hh = 12, Lp = 224;
  constexpr int Mtok = 64 * 197;   // 12608
  constexpr int Mpat = 64 * 196;   // 12544

  char* ws = (char*)d_ws;
  size_t off = 0;
  auto alloc = [&](size_t bytes) -> char* {
    char* p = ws + off;
    off += (bytes + 255) & ~(size_t)255;
    return p;
  };
  float*          x1   = (float*)alloc((size_t)Mtok * Dm * 4);
  unsigned short* xn   = (unsigned short*)alloc((size_t)Mtok * Dm * 2);
  unsigned short* qb   = (unsigned short*)alloc((size_t)64 * Hh * Lp * 64 * 2);
  unsigned short* kb   = (unsigned short*)alloc((size_t)64 * Hh * Lp * 64 * 2);
  unsigned short* vb   = (unsigned short*)alloc((size_t)64 * Hh * Lp * 64 * 2);
  unsigned short* mixb = (unsigned short*)alloc((size_t)Mtok * Dm * 2);
  char* region = alloc((size_t)Mtok * MLPD * 2);   // union: {pd,pg,pu} | h
  float*          pd = (float*)region;
  unsigned short* pg = (unsigned short*)(region + (size_t)Mtok * DHm * 4);
  float*          pu = (float*)(region + (size_t)Mtok * DHm * 4 +
                                         (size_t)Mpat * DHm * 2);
  unsigned short* hb = (unsigned short*)region;

  // 1) pd = x @ w_down + b_down  (all 12608 rows; conv selects patches)
  gemm_k<false, 0, 0, false><<<dim3(DHm / 64, (Mtok + 127) / 128), 256, 0,
      stream>>>(x, w_down, b_down, nullptr, pd, Mtok, DHm, Dm, 0);
  // 2) depthwise temporal conv + quickGELU -> pg (bf16)
  convgelu_k<<<((size_t)Mpat * DHm + 255) / 256, 256, 0, stream>>>(
      pd, w_conv, b_conv, pg);
  // 3) pu = pg @ w_up + b_up
  gemm_k<true, 0, 0, false><<<dim3(Dm / 64, (Mpat + 127) / 128), 256, 0,
      stream>>>(pg, w_up, b_up, nullptr, pu, Mpat, Dm, DHm, 0);
  // 4) x1 = x + concat([cls, pu])
  addmsg_k<<<((size_t)Mtok * Dm + 255) / 256, 256, 0, stream>>>(x, pu, x1);
  // 5) xn = LN(x1) -> bf16
  ln_k<<<Mtok, 256, 0, stream>>>(x1, g1, be1, xn);
  // 5b) zero padded q/k/v (rows t in [197,224))
  {
    size_t nw = (size_t)3 * 64 * Hh * Lp * 64 * 2 / 4;
    zero_k<<<(unsigned)((nw + 255) / 256), 256, 0, stream>>>(
        (unsigned int*)qb, nw);
  }
  // 6) q/k/v GEMMs with per-head scatter
  gemm_k<true, 0, 2, false><<<dim3(Dm / 64, (Mtok + 127) / 128), 256, 0,
      stream>>>(xn, wq, bq, nullptr, qb, Mtok, Dm, Dm, Lp);
  gemm_k<true, 0, 2, false><<<dim3(Dm / 64, (Mtok + 127) / 128), 256, 0,
      stream>>>(xn, wkp, bk, nullptr, kb, Mtok, Dm, Dm, Lp);
  gemm_k<true, 0, 2, false><<<dim3(Dm / 64, (Mtok + 127) / 128), 256, 0,
      stream>>>(xn, wvp, bv, nullptr, vb, Mtok, Dm, Dm, Lp);
  // 7) flash attention -> mix (bf16, token-major)
  attn_k<<<64 * Hh, 256, 0, stream>>>(qb, kb, vb, mixb);
  // 8) out(x2) = mix @ wo + bo + x1
  gemm_k<true, 0, 0, true><<<dim3(Dm / 64, (Mtok + 127) / 128), 256, 0,
      stream>>>(mixb, wo, bo, x1, out, Mtok, Dm, Dm, 0);
  // 9) xn = LN(x2) -> bf16
  ln_k<<<Mtok, 256, 0, stream>>>(out, g2, be2, xn);
  // 10) h = quickGELU(xn @ w1 + b1) -> bf16
  gemm_k<true, 1, 1, false><<<dim3(MLPD / 64, (Mtok + 127) / 128), 256, 0,
      stream>>>(xn, w1, b1, nullptr, hb, Mtok, MLPD, Dm, 0);
  // 11) out = h @ w2 + b2 + x2  (element-wise read-then-write of d_out)
  gemm_k<true, 0, 0, true><<<dim3(Dm / 64, (Mtok + 127) / 128), 256, 0,
      stream>>>(hb, w2, b2, out, out, Mtok, Dm, MLPD, 0);
}
